// DimenetPP_85779086836107
// MI455X (gfx1250) — matrix operand order repaired and checked by reading, not compiled, not run on hardware
//
#include <hip/hip_runtime.h>
#include <stdint.h>
#include <stddef.h>

// ============================================================================
// DimeNet++ forward for MI455X (gfx1250, wave32, WMMA).
// Dense layers -> v_wmma_f32_16x16x32_f16 (A staged in LDS as f16, B = f16
// weight copies prepared once per launch in d_ws, L2-resident ~4MB).
// K=6/K=8 basis projections folded into combined weights on device.
// Triplet path: fused angle->cbf->sbf(LDS)->WMMA(42->64, K zero-padded to 64)
// ->gather x_down[idx_kj]->f32 global atomic segment-sum by idx_ji.
// ============================================================================

typedef __attribute__((ext_vector_type(16))) _Float16 v16h;
typedef __attribute__((ext_vector_type(8)))  _Float16 v8h;
typedef __attribute__((ext_vector_type(8)))  float    v8f;
union U16h { v16h v; v8h h[2]; };

#define DEV __device__ __forceinline__

static constexpr int NEDGE = 200000;
static constexpr int NTRIP = 600000;
static constexpr int NATOM = 20000;
static constexpr int NMOLS = 128;

__constant__ float c_zeros[7][6] = {
  {3.14159265f, 6.28318531f, 9.42477796f, 12.56637061f, 15.70796327f, 18.84955592f},
  {4.49340946f, 7.72525184f, 10.90412166f, 14.06619391f, 17.22075527f, 20.37130296f},
  {5.76345920f, 9.09501133f, 12.32294097f, 15.51460301f, 18.68903635f, 21.85387422f},
  {6.98793200f, 10.41711855f, 13.69802315f, 16.92362129f, 20.12180617f, 23.30424716f},
  {8.18256145f, 11.70490715f, 15.03966471f, 18.30125596f, 21.52541763f, 24.72756553f},
  {9.35581211f, 12.96653017f, 16.35470964f, 19.65315286f, 22.90455064f, 26.12775039f},
  {10.51283540f,14.20739236f, 17.64797496f, 20.98346327f, 24.26276805f, 27.50786836f}};

__constant__ float c_cbf[7] = {0.28209479f,0.48860251f,0.63078313f,0.74635267f,
                               0.84628438f,0.93560258f,1.01710723f};

DEV float swishf(float x){ return x / (1.f + __expf(-x)); }

// ---- WMMA 16xNOUT tile over K=KIN, A in LDS [16][KIN] f16, B row-major f16 ----
template<int KIN>
DEV v8f wmma_tile(const _Float16 (*As)[KIN], const _Float16* __restrict__ Wrow, int lane){
  const int rowA = lane & 15;
  const int kbA  = (lane >> 4) * 8;    // A: half-wave K split 0..7 / 8..15 (+16 in h[1])
  const int kbB  = (lane >> 4) * 8;    // B: same K split as A: 0..7 / 8..15 (+16 in h[1])
  const _Float16* arow = As[rowA];
  v8f acc = {};
#pragma unroll
  for (int k0 = 0; k0 < KIN; k0 += 32) {
    U16h a, b;
    a.h[0] = *(const v8h*)(arow + k0 + kbA);
    a.h[1] = *(const v8h*)(arow + k0 + kbA + 16);
    b.h[0] = *(const v8h*)(Wrow + k0 + kbB);
    b.h[1] = *(const v8h*)(Wrow + k0 + kbB + 16);
    acc = __builtin_amdgcn_wmma_f32_16x16x32_f16(false, a.v, false, b.v,
                                                 (short)0, acc, false, false);
  }
  return acc;
}

// ---------------------------------------------------------------------------
// prep kernels
// ---------------------------------------------------------------------------
__global__ void cvt_kernel(const float* __restrict__ src, _Float16* __restrict__ dst, int n){
  int i = blockIdx.x*256 + threadIdx.x;
  if (i < n) dst[i] = (_Float16)src[i];
}

__global__ void norm_kernel(float* __restrict__ out){  // 1/sqrt(0.5*j_{l+1}(z)^2)
  int i = threadIdx.x; if (i >= 42) return;
  int l = i/6, k = i%6;
  float z = c_zeros[l][k];
  float s = __sinf(z), c = __cosf(z);
  float jm2 = s/z, jm1 = s/(z*z) - c/z;    // j0, j1
  for (int m = 2; m <= l+1; ++m){ float t = (2.f*m-1.f)/z*jm1 - jm2; jm2=jm1; jm1=t; }
  out[i] = rsqrtf(0.5f*jm1*jm1);
}

__global__ void combine_rbf_kernel(const float* __restrict__ W1 /*[8][6]*/,
                                   const float* __restrict__ W2 /*[128][8]*/,
                                   float* __restrict__ out /*[128][6]*/){
  int i = blockIdx.x*256 + threadIdx.x; if (i >= 128*6) return;
  int c = i/6, k = i%6;
  float s = 0.f;
  for (int b = 0; b < 8; ++b) s += W2[c*8+b]*W1[b*6+k];
  out[i] = s;
}

__global__ void combine_sbf_kernel(const float* __restrict__ W1 /*[8][42]*/,
                                   const float* __restrict__ W2 /*[64][8]*/,
                                   _Float16* __restrict__ out /*[64][64] K-padded*/){
  int i = blockIdx.x*256 + threadIdx.x; if (i >= 64*64) return;
  int o = i >> 6, j = i & 63;
  float s = 0.f;
  if (j < 42) for (int b = 0; b < 8; ++b) s += W2[o*8+b]*W1[b*42+j];
  out[i] = (_Float16)s;
}

// ---------------------------------------------------------------------------
// stage 1: per-edge rbf[E,6] and srbf[E,42] (env folded in)
// ---------------------------------------------------------------------------
__global__ void edge_kernel(const float* __restrict__ R, const int* __restrict__ ei,
                            const int* __restrict__ ej, const float* __restrict__ freq,
                            const float* __restrict__ snorm,
                            float* __restrict__ rbf, float* __restrict__ srbf, int E){
  int e = blockIdx.x*256 + threadIdx.x; if (e >= E) return;
  int i = ei[e], j = ej[e];
  float dx = R[i*3+0]-R[j*3+0], dy = R[i*3+1]-R[j*3+1], dz = R[i*3+2]-R[j*3+2];
  float d = sqrtf(fmaxf(dx*dx+dy*dy+dz*dz, 0.f)) * (1.f/5.0f);
  d = fmaxf(d, 1e-12f);
  float d5 = d*d; d5 *= d5; d5 *= d;    // d^5
  float env = (d < 1.f) ? (1.f/d - 28.f*d5 + 48.f*d5*d - 21.f*d5*d*d) : 0.f;
  for (int k = 0; k < 6; ++k) rbf[(size_t)e*6+k] = env*__sinf(freq[k]*d);
  for (int l = 0; l < 7; ++l){
    for (int k = 0; k < 6; ++k){
      float x = c_zeros[l][k]*d;
      float s = __sinf(x), c = __cosf(x);
      float jl = s/x;
      if (l >= 1){
        float jm2 = jl, jm1 = s/(x*x) - c/x;
        for (int m = 2; m <= l; ++m){ float t=(2.f*m-1.f)/x*jm1 - jm2; jm2=jm1; jm1=t; }
        jl = jm1;
      }
      srbf[(size_t)e*42 + l*6 + k] = env*snorm[l*6+k]*jl;
    }
  }
}

// ---------------------------------------------------------------------------
// embedding: x = swish([h_i | h_j | h_rbf] @ emb_W.T + emb_b), K=384 WMMA
// ---------------------------------------------------------------------------
__global__ __launch_bounds__(256)
void embed_kernel(const float* __restrict__ emb_table, const int* __restrict__ Zt,
                  const int* __restrict__ ei, const int* __restrict__ ej,
                  const float* __restrict__ rbf, const float* __restrict__ erbfW,
                  const float* __restrict__ erbfB, const _Float16* __restrict__ W,
                  const float* __restrict__ bias, float* __restrict__ out, int M){
  __shared__ __align__(16) _Float16 As[16][384];
  __shared__ int zi[16], zj[16];
  int m0 = blockIdx.x*16, tid = threadIdx.x;
  if (tid < 16){ int e = m0+tid; zi[tid] = Zt[ei[e]]; zj[tid] = Zt[ej[e]]; }
  __syncthreads();
  for (int i = tid; i < 16*384; i += 256){
    int r = i/384, c = i%384, e = m0+r;
    float v;
    if (c < 128)      v = emb_table[(size_t)zi[r]*128 + c];
    else if (c < 256) v = emb_table[(size_t)zj[r]*128 + (c-128)];
    else {
      int cc = c-256;
      float s = erbfB[cc];
      for (int k = 0; k < 6; ++k) s += rbf[(size_t)e*6+k]*erbfW[cc*6+k];
      v = swishf(s);
    }
    As[r][c] = (_Float16)v;
  }
  __syncthreads();
  int wave = tid >> 5, lane = tid & 31;
  int n0 = wave*16;
  v8f acc = wmma_tile<384>(As, W + (size_t)(n0 + (lane&15))*384, lane);
  int nc = n0 + (lane & 15);
  int mb = (lane >> 4)*8;
  float bv = bias[nc];
#pragma unroll
  for (int r = 0; r < 8; ++r){
    int gm = m0 + mb + r;
    if (gm < M) out[(size_t)gm*128 + nc] = swishf(acc[r] + bv);
  }
}

// ---------------------------------------------------------------------------
// generic GEMM:  out = epilogue(A @ W.T)
// ---------------------------------------------------------------------------
constexpr int EPI_NONE = 0, EPI_SWISH = 1, EPI_SWISH_RBF = 2, EPI_SWISH_ADD = 3;

template<int KIN,int NOUT,int MT,int NTW,int EPI,bool HASBIAS>
__global__ __launch_bounds__(32*MT*NTW)
void gemm_kernel(const float* __restrict__ A, const _Float16* __restrict__ W,
                 const float* __restrict__ bias, const float* __restrict__ add,
                 const float* __restrict__ rbf, const float* __restrict__ Wc,
                 float* __restrict__ out, int M){
  constexpr int ROWS = 16*MT, NTH = 32*MT*NTW;
  __shared__ __align__(16) _Float16 As[ROWS][KIN];
  int m0 = blockIdx.x*ROWS, tid = threadIdx.x;
  for (int i = tid; i < ROWS*KIN; i += NTH){
    int r = i/KIN, c = i%KIN;
    float v = (m0 + r < M) ? A[(size_t)(m0+r)*KIN + c] : 0.f;
    As[r][c] = (_Float16)v;
  }
  __syncthreads();
  int wave = tid >> 5, lane = tid & 31;
  int mt = wave / NTW, nt = wave % NTW;
  int n0 = nt*16;
  v8f acc = wmma_tile<KIN>(&As[mt*16], W + (size_t)(n0 + (lane&15))*KIN, lane);
  int nc = n0 + (lane & 15);
  int mb = mt*16 + (lane >> 4)*8;
  float bv = HASBIAS ? bias[nc] : 0.f;
  float wc[6];
  if (EPI == EPI_SWISH_RBF){
#pragma unroll
    for (int k = 0; k < 6; ++k) wc[k] = Wc[nc*6+k];
  }
#pragma unroll
  for (int r = 0; r < 8; ++r){
    int gm = m0 + mb + r;
    if (gm >= M) continue;
    float v = acc[r] + bv;
    if (EPI == EPI_SWISH) v = swishf(v);
    else if (EPI == EPI_SWISH_ADD) v = swishf(v) + add[(size_t)gm*NOUT + nc];
    else if (EPI == EPI_SWISH_RBF){
      float s = 0.f;
#pragma unroll
      for (int k = 0; k < 6; ++k) s += rbf[(size_t)gm*6+k]*wc[k];
      v = swishf(v)*s;
    }
    out[(size_t)gm*NOUT + nc] = v;
  }
}

// ---------------------------------------------------------------------------
// fused triplet kernel: angle -> cbf, sbf tile in LDS, WMMA (42->64, K pad 64),
// multiply gathered x_down[idx_kj], atomic segment-sum by idx_ji into acc[E,64]
// ---------------------------------------------------------------------------
__global__ __launch_bounds__(256)
void triplet_kernel(const float* __restrict__ R, const int* __restrict__ i3,
                    const int* __restrict__ j3, const int* __restrict__ k3,
                    const int* __restrict__ idxkj, const int* __restrict__ idxji,
                    const float* __restrict__ srbf, const _Float16* __restrict__ Wc,
                    const float* __restrict__ down, float* __restrict__ accb, int T){
  __shared__ __align__(16) _Float16 As[32][64];
  __shared__ float cbf_s[32][7];
  __shared__ int kj_s[32], ji_s[32];
  int t0 = blockIdx.x*32, tid = threadIdx.x;
  if (tid < 32){
    int t = t0 + tid;
    int ia = i3[t]*3, ja = j3[t]*3, ka = k3[t]*3;
    float r1x = R[ja]-R[ia],   r1y = R[ja+1]-R[ia+1], r1z = R[ja+2]-R[ia+2];
    float r2x = R[ka]-R[ja],   r2y = R[ka+1]-R[ja+1], r2z = R[ka+2]-R[ja+2];
    float cx = r1y*r2z - r1z*r2y, cy = r1z*r2x - r1x*r2z, cz = r1x*r2y - r1y*r2x;
    float cn2 = cx*cx + cy*cy + cz*cz;
    float dp  = r1x*r2x + r1y*r2y + r1z*r2z;
    float z = dp * rsqrtf(dp*dp + cn2 + 1e-30f);   // cos(atan2(|cross|, dot))
    float p0 = 1.f, p1 = z;
    cbf_s[tid][0] = c_cbf[0];
    cbf_s[tid][1] = z*c_cbf[1];
    for (int l = 2; l < 7; ++l){
      float p = ((2.f*l-1.f)*z*p1 - (l-1.f)*p0)/(float)l;
      p0 = p1; p1 = p;
      cbf_s[tid][l] = p*c_cbf[l];
    }
    kj_s[tid] = idxkj[t]; ji_s[tid] = idxji[t];
  }
  __syncthreads();
  for (int i = tid; i < 32*64; i += 256){
    int r = i >> 6, c = i & 63;
    float v = (c < 42) ? srbf[(size_t)kj_s[r]*42 + c]*cbf_s[r][c/6] : 0.f;
    As[r][c] = (_Float16)v;
  }
  __syncthreads();
  int wave = tid >> 5, lane = tid & 31;
  int mt = wave >> 2, nt = wave & 3;      // 2 row-tiles x 4 col-tiles
  int n0 = nt*16;
  v8f acc = wmma_tile<64>(&As[mt*16], Wc + (size_t)(n0 + (lane&15))*64, lane);
  int nc = n0 + (lane & 15);
  int mb = mt*16 + (lane >> 4)*8;
#pragma unroll
  for (int r = 0; r < 8; ++r){
    int m = mb + r;
    if (t0 + m >= T) continue;
    float v = acc[r] * down[(size_t)kj_s[m]*64 + nc];
    atomicAdd(&accb[(size_t)ji_s[m]*64 + nc], v);
  }
}

// ---------------------------------------------------------------------------
// output-block helpers
// ---------------------------------------------------------------------------
__global__ __launch_bounds__(128)
void outscatter_kernel(const float* __restrict__ x, const float* __restrict__ rbf,
                       const float* __restrict__ Wrbf /*[128][6] f32*/,
                       const int* __restrict__ ei, float* __restrict__ nodeg, int E){
  int e = blockIdx.x; if (e >= E) return;
  int c = threadIdx.x;
  const float* rr = rbf + (size_t)e*6;
  const float* wr = Wrbf + c*6;
  float s = 0.f;
#pragma unroll
  for (int k = 0; k < 6; ++k) s += rr[k]*wr[k];
  atomicAdd(&nodeg[(size_t)ei[e]*128 + c], s * x[(size_t)e*128 + c]);
}

__global__ void wout_kernel(const float* __restrict__ g, const float* __restrict__ Wout,
                            float* __restrict__ Pn, int N){
  int nd = blockIdx.x*256 + threadIdx.x; if (nd >= N) return;
  float s = 0.f;
  for (int c = 0; c < 256; ++c) s += g[(size_t)nd*256 + c]*Wout[c];
  Pn[nd] += s;
}

__global__ void molsum_kernel(const float* __restrict__ Pn, const int* __restrict__ seg,
                              float* __restrict__ out, int N){
  int nd = blockIdx.x*256 + threadIdx.x; if (nd >= N) return;
  atomicAdd(&out[seg[nd]], Pn[nd]);
}

// ===========================================================================
// host side
// ===========================================================================
enum { L_EMB_W=0, L_EMB_B, L_EMB_RBF_W, L_EMB_RBF_B, L_EMB_TABLE, L_FREQ,
       L_INT0 = 6, L_OUT0 = 6 + 4*24, NLEAF = 6 + 4*24 + 5*9 };
enum { IB_W_DOWN=0, IB_W_JI, IB_W_KJ, IB_W_RBF1, IB_W_RBF2, IB_W_SBF1, IB_W_SBF2,
       IB_W_SKIP, IB_W_UP, IB_A0W1, IB_A0B1, IB_A0W2, IB_A0B2,
       IB_A1W1, IB_A1B1, IB_A1W2, IB_A1B2, IB_B_JI, IB_B_KJ, IB_B_SKIP,
       IB_BW1, IB_BB1, IB_BW2, IB_BB2 };
enum { OB_W_OUT=0, OB_W_RBF, OB_W_UP, OB_L0W, OB_L0B, OB_L1W, OB_L1B, OB_L2W, OB_L2B };

static void build_leaves(void* const* d_in, int n_in, const float** leaf){
  int sz[NLEAF]; int p = 0;
  sz[p++]=128*384; sz[p++]=128; sz[p++]=128*6; sz[p++]=128; sz[p++]=95*128; sz[p++]=6;
  for (int b = 0; b < 4; ++b){
    const int s[24] = {64*128,128*128,128*128,8*6,128*8,8*42,64*8,128*128,128*64,
                       128*128,128,128*128,128, 128*128,128,128*128,128,
                       128,128,128, 128*128,128,128*128,128};
    for (int i = 0; i < 24; ++i) sz[p++] = s[i];
  }
  for (int o = 0; o < 5; ++o){
    const int s[9] = {256,128*6,256*128,256*256,256,256*256,256,256*256,256};
    for (int i = 0; i < 9; ++i) sz[p++] = s[i];
  }
  if (n_in >= 10 + NLEAF){                  // params passed as individual leaves
    for (int i = 0; i < NLEAF; ++i) leaf[i] = (const float*)d_in[10+i];
  } else {                                   // params passed as one flat f32 blob
    const float* base = (const float*)d_in[10];
    size_t off = 0;
    for (int i = 0; i < NLEAF; ++i){ leaf[i] = base + off; off += sz[i]; }
  }
}

extern "C" void kernel_launch(void* const* d_in, const int* in_sizes, int n_in,
                              void* d_out, int out_size, void* d_ws, size_t ws_size,
                              hipStream_t stream){
  (void)in_sizes; (void)ws_size;
  const float* R       = (const float*)d_in[1];
  const int*   Zt      = (const int*)d_in[0];
  const int*   bseg    = (const int*)d_in[2];
  const int*   ei      = (const int*)d_in[3];
  const int*   ej      = (const int*)d_in[4];
  const int*   idkj    = (const int*)d_in[5];
  const int*   idji    = (const int*)d_in[6];
  const int*   i3      = (const int*)d_in[7];
  const int*   j3      = (const int*)d_in[8];
  const int*   k3      = (const int*)d_in[9];
  const float* leaf[NLEAF];
  build_leaves(d_in, n_in, leaf);

  // ---- workspace bump allocator ----
  uintptr_t wp = (uintptr_t)d_ws;
  auto alloc = [&](size_t bytes)->void*{
    void* p = (void*)wp; wp += (bytes + 255) & ~(size_t)255; return p;
  };
  const size_t E = NEDGE, N = NATOM, T = NTRIP;
  float* srbf  = (float*)alloc(E*42*4);
  float* rbf   = (float*)alloc(E*6*4);
  float* x     = (float*)alloc(E*128*4);
  float* bufA  = (float*)alloc(E*128*4);
  float* bufB  = (float*)alloc(E*128*4);
  float* bufC  = (float*)alloc(E*128*4);
  float* down  = (float*)alloc(E*64*4);
  float* accb  = (float*)alloc(E*64*4);
  float* nodeg = (float*)alloc(N*128*4);
  float* nodeA = (float*)alloc(N*256*4);
  float* nodeB = (float*)alloc(N*256*4);
  float* Pn    = (float*)alloc(N*4);
  float* snorm = (float*)alloc(42*4);
  _Float16* embW16 = (_Float16*)alloc((size_t)128*384*2);
  _Float16 *ji16[4],*kj16[4],*dn16[4],*up16[4],*sk16[4],*bw1[4],*bw2[4],
           *a0w1[4],*a0w2[4],*a1w1[4],*a1w2[4],*sbfc[4];
  float* wcr[4];
  for (int b = 0; b < 4; ++b){
    ji16[b]=(_Float16*)alloc(16384*2); kj16[b]=(_Float16*)alloc(16384*2);
    dn16[b]=(_Float16*)alloc(8192*2);  up16[b]=(_Float16*)alloc(8192*2);
    sk16[b]=(_Float16*)alloc(16384*2); bw1[b]=(_Float16*)alloc(16384*2);
    bw2[b]=(_Float16*)alloc(16384*2);  a0w1[b]=(_Float16*)alloc(16384*2);
    a0w2[b]=(_Float16*)alloc(16384*2); a1w1[b]=(_Float16*)alloc(16384*2);
    a1w2[b]=(_Float16*)alloc(16384*2); sbfc[b]=(_Float16*)alloc(4096*2);
    wcr[b]=(float*)alloc(768*4);
  }
  _Float16 *oup16[5], *ol16[5][3];
  for (int o = 0; o < 5; ++o){
    oup16[o]=(_Float16*)alloc((size_t)256*128*2);
    for (int l = 0; l < 3; ++l) ol16[o][l]=(_Float16*)alloc((size_t)256*256*2);
  }

  auto cvt = [&](const float* s, _Float16* d, int n){
    cvt_kernel<<<(n+255)/256, 256, 0, stream>>>(s, d, n);
  };

  // ---- prep: f16 weights, combined basis weights, sbf norms ----
  norm_kernel<<<1, 64, 0, stream>>>(snorm);
  cvt(leaf[L_EMB_W], embW16, 128*384);
  for (int b = 0; b < 4; ++b){
    const float* const* lb = leaf + L_INT0 + b*24;
    cvt(lb[IB_W_JI],  ji16[b], 16384);  cvt(lb[IB_W_KJ],  kj16[b], 16384);
    cvt(lb[IB_W_DOWN],dn16[b], 8192);   cvt(lb[IB_W_UP],  up16[b], 8192);
    cvt(lb[IB_W_SKIP],sk16[b], 16384);
    cvt(lb[IB_BW1],   bw1[b],  16384);  cvt(lb[IB_BW2],   bw2[b],  16384);
    cvt(lb[IB_A0W1],  a0w1[b], 16384);  cvt(lb[IB_A0W2],  a0w2[b], 16384);
    cvt(lb[IB_A1W1],  a1w1[b], 16384);  cvt(lb[IB_A1W2],  a1w2[b], 16384);
    combine_rbf_kernel<<<3, 256, 0, stream>>>(lb[IB_W_RBF1], lb[IB_W_RBF2], wcr[b]);
    combine_sbf_kernel<<<16, 256, 0, stream>>>(lb[IB_W_SBF1], lb[IB_W_SBF2], sbfc[b]);
  }
  for (int o = 0; o < 5; ++o){
    const float* const* lo = leaf + L_OUT0 + o*9;
    cvt(lo[OB_W_UP], oup16[o], 256*128);
    cvt(lo[OB_L0W], ol16[o][0], 256*256);
    cvt(lo[OB_L1W], ol16[o][1], 256*256);
    cvt(lo[OB_L2W], ol16[o][2], 256*256);
  }

  hipMemsetAsync(Pn, 0, N*4, stream);
  hipMemsetAsync(d_out, 0, (size_t)out_size*4, stream);

  // ---- stage 1: edge bases ----
  edge_kernel<<<(NEDGE+255)/256, 256, 0, stream>>>(R, ei, ej, leaf[L_FREQ], snorm,
                                                   rbf, srbf, NEDGE);
  // ---- embedding ----
  embed_kernel<<<NEDGE/16, 256, 0, stream>>>(leaf[L_EMB_TABLE], Zt, ei, ej, rbf,
                                             leaf[L_EMB_RBF_W], leaf[L_EMB_RBF_B],
                                             embW16, leaf[L_EMB_B], x, NEDGE);

  auto run_output = [&](int o, const float* xin){
    const float* const* lo = leaf + L_OUT0 + o*9;
    hipMemsetAsync(nodeg, 0, N*128*4, stream);
    outscatter_kernel<<<NEDGE, 128, 0, stream>>>(xin, rbf, lo[OB_W_RBF], ei, nodeg, NEDGE);
    gemm_kernel<128,256,1,16,EPI_NONE,false><<<NATOM/16, 512, 0, stream>>>(
        nodeg, oup16[o], nullptr, nullptr, nullptr, nullptr, nodeA, NATOM);
    gemm_kernel<256,256,1,16,EPI_SWISH,true><<<NATOM/16, 512, 0, stream>>>(
        nodeA, ol16[o][0], lo[OB_L0B], nullptr, nullptr, nullptr, nodeB, NATOM);
    gemm_kernel<256,256,1,16,EPI_SWISH,true><<<NATOM/16, 512, 0, stream>>>(
        nodeB, ol16[o][1], lo[OB_L1B], nullptr, nullptr, nullptr, nodeA, NATOM);
    gemm_kernel<256,256,1,16,EPI_SWISH,true><<<NATOM/16, 512, 0, stream>>>(
        nodeA, ol16[o][2], lo[OB_L2B], nullptr, nullptr, nullptr, nodeB, NATOM);
    wout_kernel<<<(NATOM+255)/256, 256, 0, stream>>>(nodeB, lo[OB_W_OUT], Pn, NATOM);
  };

  run_output(0, x);

  for (int b = 0; b < 4; ++b){
    const float* const* lb = leaf + L_INT0 + b*24;
    // x_ji = swish(x@W_ji + b_ji)
    gemm_kernel<128,128,1,8,EPI_SWISH,true><<<NEDGE/16, 256, 0, stream>>>(
        x, ji16[b], lb[IB_B_JI], nullptr, nullptr, nullptr, bufA, NEDGE);
    // x_kj = swish(x@W_kj + b_kj) * (rbf @ Wc_rbf.T)
    gemm_kernel<128,128,1,8,EPI_SWISH_RBF,true><<<NEDGE/16, 256, 0, stream>>>(
        x, kj16[b], lb[IB_B_KJ], nullptr, rbf, wcr[b], bufB, NEDGE);
    // down = swish(x_kj @ W_down)
    gemm_kernel<128,64,2,4,EPI_SWISH,false><<<NEDGE/32, 256, 0, stream>>>(
        bufB, dn16[b], nullptr, nullptr, nullptr, nullptr, down, NEDGE);
    // triplet message + segment sum
    hipMemsetAsync(accb, 0, E*64*4, stream);
    triplet_kernel<<<NTRIP/32, 256, 0, stream>>>(R, i3, j3, k3, idkj, idji,
                                                 srbf, sbfc[b], down, accb, NTRIP);
    // h = swish(acc @ W_up) + x_ji
    gemm_kernel<64,128,1,8,EPI_SWISH_ADD,false><<<NEDGE/16, 256, 0, stream>>>(
        accb, up16[b], nullptr, bufA, nullptr, nullptr, bufB, NEDGE);
    // residual "before"
    gemm_kernel<128,128,1,8,EPI_SWISH,true><<<NEDGE/16, 256, 0, stream>>>(
        bufB, bw1[b], lb[IB_BB1], nullptr, nullptr, nullptr, bufC, NEDGE);
    gemm_kernel<128,128,1,8,EPI_SWISH_ADD,true><<<NEDGE/16, 256, 0, stream>>>(
        bufC, bw2[b], lb[IB_BB2], bufB, nullptr, nullptr, bufA, NEDGE);
    // skip: swish(h @ W_skip + b_skip) + x
    gemm_kernel<128,128,1,8,EPI_SWISH_ADD,true><<<NEDGE/16, 256, 0, stream>>>(
        bufA, sk16[b], lb[IB_B_SKIP], x, nullptr, nullptr, bufB, NEDGE);
    // residual "after" x2  (final write back into x)
    gemm_kernel<128,128,1,8,EPI_SWISH,true><<<NEDGE/16, 256, 0, stream>>>(
        bufB, a0w1[b], lb[IB_A0B1], nullptr, nullptr, nullptr, bufC, NEDGE);
    gemm_kernel<128,128,1,8,EPI_SWISH_ADD,true><<<NEDGE/16, 256, 0, stream>>>(
        bufC, a0w2[b], lb[IB_A0B2], bufB, nullptr, nullptr, bufA, NEDGE);
    gemm_kernel<128,128,1,8,EPI_SWISH,true><<<NEDGE/16, 256, 0, stream>>>(
        bufA, a1w1[b], lb[IB_A1B1], nullptr, nullptr, nullptr, bufC, NEDGE);
    gemm_kernel<128,128,1,8,EPI_SWISH_ADD,true><<<NEDGE/16, 256, 0, stream>>>(
        bufC, a1w2[b], lb[IB_A1B2], bufA, nullptr, nullptr, x, NEDGE);

    run_output(b+1, x);
  }

  // ---- molecule reduction ----
  molsum_kernel<<<(NATOM+255)/256, 256, 0, stream>>>(Pn, bseg, (float*)d_out, NATOM);
}
